// MultiheadAdditiveAttention_3040836846165
// MI455X (gfx1250) — matrix operand order repaired and checked by reading, not compiled, not run on hardware
//
#include <hip/hip_runtime.h>

// ---------------------------------------------------------------------------
// MultiheadAdditiveAttention for MI455X (gfx1250, wave32, WMMA, async-to-LDS).
// B=1, T=S=512, E=768, H=12, D=64.
//   pass 0: f32->f16 convert of x and W* (enables byte-exact async staging)
//   q,k,v = x@W^T+b          -> WMMA f16 GEMM, async double-buffered tiles
//   score = tanh(q+k)*ws+bs  -> VALU fast tanh (v_exp_f32 + v_rcp_f32)
//   softmax over s           -> half-wave shuffle reduction
//   attn  = p @ v            -> WMMA f16 GEMM
//   out   = attn@Wo^T + bo   -> WMMA f16 GEMM
// ---------------------------------------------------------------------------

typedef __attribute__((ext_vector_type(16))) _Float16 v16h;
typedef __attribute__((ext_vector_type(8)))  _Float16 v8h;
typedef __attribute__((ext_vector_type(4)))  _Float16 v4h;
typedef __attribute__((ext_vector_type(8)))  float    v8f;

constexpr int kE = 768;
constexpr int kT = 512;
constexpr int kH = 12;
constexpr int kD = 64;

// --- CDNA5 async global->LDS copy (16B per lane), ASYNCcnt-tracked ----------
__device__ inline unsigned lds_addr32(const void* p) {
  return (unsigned)(unsigned long long)(uintptr_t)p;  // low 32 bits = LDS offset
}
__device__ inline void async_cp16(unsigned ldsoff, const void* g) {
  asm volatile("global_load_async_to_lds_b128 %0, %1, off"
               :: "v"(ldsoff), "v"((unsigned long long)(uintptr_t)g)
               : "memory");
}
__device__ inline void wait_async0() {
  asm volatile("s_wait_asynccnt 0x0" ::: "memory");
}

// --- WMMA fragment packing per CDNA5 ISA 7.12.2 (wave32) --------------------
// A (16x32 f16 tile, row-major): lane l<16 -> row M=l; halves 0..7 = K[8hi..],
// halves 8..15 = K[16+8hi..]  (hi = lane>>4).
__device__ inline v16h load_frag_a(const _Float16* __restrict__ row, int hi) {
  union { v16h v; v8h h[2]; } u;
  u.h[0] = *(const v8h*)(row + 8 * hi);
  u.h[1] = *(const v8h*)(row + 16 + 8 * hi);
  return u.v;
}
// B (32x16) staged TRANSPOSED: colbase = &ldsBT[n][0]; halves 0..7 = K[8hi..], halves 8..15 = K[16+8hi..].
__device__ inline v16h load_frag_b(const _Float16* __restrict__ colbase, int hi) {
  union { v16h v; v8h h[2]; } u;
  u.h[0] = *(const v8h*)(colbase + 8 * hi);
  u.h[1] = *(const v8h*)(colbase + 16 + 8 * hi);
  return u.v;
}

__device__ inline float fast_tanh(float x) {
  // tanh(x) = 1 - 2/(exp2(2x*log2e)+1): v_exp_f32 + v_rcp_f32
  float e = __builtin_amdgcn_exp2f(x * 2.8853900817779268f);
  return 1.0f - 2.0f * __builtin_amdgcn_rcpf(e + 1.0f);
}

// ---------------------------------------------------------------------------
// Pass 0: vectorized f32 -> f16 conversion (x and the four weight matrices).
// ---------------------------------------------------------------------------
__global__ __launch_bounds__(256) void cvt_f32_f16(const float* __restrict__ s,
                                                   _Float16* __restrict__ d,
                                                   int n4) {
  int i = blockIdx.x * 256 + threadIdx.x;
  if (i < n4) {
    float4 f = ((const float4*)s)[i];
    v4h h = {(_Float16)f.x, (_Float16)f.y, (_Float16)f.z, (_Float16)f.w};
    ((v4h*)d)[i] = h;
  }
}

// ---------------------------------------------------------------------------
// out[m,n] = sum_k A[m,k]*W[n,k] + bias[n]  (A, W pre-converted to f16).
// grid: (512/128, 768/64); block 256 = 8 waves.
// Block tile 128(M) x 64(N); wave w: N-stripe (w&3)*16, M-half (w>>2)*64
// -> 4 WMMA / wave / K-step, B fragment reused. Tiles async double-buffered.
// ---------------------------------------------------------------------------
__global__ __launch_bounds__(256) void gemm_h_xwt_bias(
    const _Float16* __restrict__ Ah, const _Float16* __restrict__ Wh,
    const float* __restrict__ bias, float* __restrict__ out) {
  __shared__ alignas(16) _Float16 ldsA[2][128][40];  // padded pitch (80B rows)
  __shared__ alignas(16) _Float16 ldsB[2][64][40];   // ldsB[n][k] = W[n][k0+k]

  const int tid   = threadIdx.x;
  const int lane  = tid & 31;
  const int w     = tid >> 5;
  const int l     = lane & 15;
  const int hi    = lane >> 4;
  const int mBase = blockIdx.x * 128;
  const int nTile = blockIdx.y * 64;
  const int nSub  = (w & 3) * 16;
  const int mSub  = (w >> 2) * 64;

  auto stage = [&](int buf, int k0) {
    // A tile: 128 rows x 64B = 512 chunks of 16B -> 2 per thread
#pragma unroll
    for (int t = 0; t < 2; ++t) {
      int c = tid + t * 256, r = c >> 2, c16 = c & 3;
      async_cp16(lds_addr32(&ldsA[buf][r][c16 * 8]),
                 Ah + (mBase + r) * kE + k0 + c16 * 8);
    }
    // B tile: 64 rows x 64B = 256 chunks -> 1 per thread
    {
      int r = tid >> 2, c16 = tid & 3;
      async_cp16(lds_addr32(&ldsB[buf][r][c16 * 8]),
                 Wh + (nTile + r) * kE + k0 + c16 * 8);
    }
  };

  v8f acc[4] = {};
  stage(0, 0);
  wait_async0();
  __syncthreads();
  for (int k0 = 0; k0 < kE; k0 += 32) {
    const int buf = (k0 >> 5) & 1;
    if (k0 + 32 < kE) stage(buf ^ 1, k0 + 32);  // overlap with compute
    v16h b = load_frag_b(&ldsB[buf][nSub + l][0], hi);
#pragma unroll
    for (int i = 0; i < 4; ++i) {
      v16h a = load_frag_a(&ldsA[buf][mSub + 16 * i + l][0], hi);
      acc[i] = __builtin_amdgcn_wmma_f32_16x16x32_f16(false, a, false, b,
                                                      (short)0, acc[i], false,
                                                      false);
    }
    wait_async0();
    __syncthreads();
  }

  const int nGlob = nTile + nSub + l;
  const float bv  = bias[nGlob];
#pragma unroll
  for (int i = 0; i < 4; ++i)
#pragma unroll
    for (int r = 0; r < 8; ++r) {
      const int m = mBase + mSub + 16 * i + hi * 8 + r;
      out[m * kE + nGlob] = acc[i][r] + bv;
    }
}

// ---------------------------------------------------------------------------
// Additive attention core: scores + softmax + p@v per (head, 16-row t-tile).
// grid: (T/16, H); block 256 (8 waves). k-tiles async double-buffered (f32).
// Writes attn directly as f16 for the output-projection GEMM.
// ---------------------------------------------------------------------------
__global__ __launch_bounds__(256) void attn_kernel(
    const float* __restrict__ Qb, const float* __restrict__ Kb,
    const float* __restrict__ Vb, const float* __restrict__ wsp,
    const float* __restrict__ bsp, _Float16* __restrict__ attnH) {
  __shared__ alignas(16) float    qtile[16][kD + 4];
  __shared__ alignas(16) float    ktile[2][16][kD + 4];
  __shared__ alignas(16) float    wsl[kD];
  __shared__ alignas(16) float    sc[16][kT + 4];   // scores -> exp values
  __shared__ alignas(16) _Float16 pH[16][520];      // f16 probabilities
  __shared__ alignas(16) _Float16 vT[kD][40];       // vT[d][s_rel]

  const int tid   = threadIdx.x;
  const int tBase = blockIdx.x * 16;
  const int h     = blockIdx.y;
  const int ty    = tid >> 4;  // t row (shared by half-wave)
  const int tx    = tid & 15;

  {  // async stage q tile (16x64 f32) + first k chunk
    const int r  = tid >> 4;
    const int c4 = (tid & 15) * 4;
    async_cp16(lds_addr32(&qtile[r][c4]), Qb + (tBase + r) * kE + h * kD + c4);
    async_cp16(lds_addr32(&ktile[0][r][c4]), Kb + r * kE + h * kD + c4);
  }
  if (tid < kD) wsl[tid] = wsp[tid];
  wait_async0();
  __syncthreads();

  const float bscale = bsp[0];
  const float L2E    = 1.4426950408889634f;

  // ---- phase 1: scores (201M tanh total across grid) ----
  for (int s0 = 0; s0 < kT; s0 += 16) {
    const int buf = (s0 >> 4) & 1;
    if (s0 + 16 < kT) {  // prefetch next k chunk while computing
      const int r  = tid >> 4;
      const int c4 = (tid & 15) * 4;
      async_cp16(lds_addr32(&ktile[buf ^ 1][r][c4]),
                 Kb + (s0 + 16 + r) * kE + h * kD + c4);
    }
    float sum = 0.f;
#pragma unroll 8
    for (int d = 0; d < kD; ++d)
      sum = fmaf(fast_tanh(qtile[ty][d] + ktile[buf][tx][d]), wsl[d], sum);
    sc[ty][s0 + tx] = sum + bscale;
    wait_async0();
    __syncthreads();
  }

  // ---- phase 2: row softmax (16 lanes per row, xor-shuffle reduce) ----
  float m = -1e30f;
  for (int j = tx; j < kT; j += 16) m = fmaxf(m, sc[ty][j]);
#pragma unroll
  for (int o = 8; o > 0; o >>= 1) m = fmaxf(m, __shfl_xor(m, o, 32));
  float s = 0.f;
  for (int j = tx; j < kT; j += 16) {
    float e = __builtin_amdgcn_exp2f((sc[ty][j] - m) * L2E);
    sc[ty][j] = e;
    s += e;
  }
#pragma unroll
  for (int o = 8; o > 0; o >>= 1) s += __shfl_xor(s, o, 32);
  const float inv = __builtin_amdgcn_rcpf(s);
  for (int j = tx; j < kT; j += 16) pH[ty][j] = (_Float16)(sc[ty][j] * inv);
  __syncthreads();

  // ---- phase 3: attn(16x64) = p(16x512) @ v(512x64) via WMMA ----
  const int wv = tid >> 5;  // waves 0..3 own d-subtiles
  const int l  = tid & 15;
  const int hi = (tid >> 4) & 1;
  v8f acc = {};
  for (int kc = 0; kc < kT; kc += 32) {
    {  // stage v chunk transposed+converted: vT[d][srel] = V[kc+srel][h*64+d]
      const int srel = tid >> 3;
      const int c0   = (tid & 7) * 8;
      const float* src = Vb + (kc + srel) * kE + h * kD + c0;
#pragma unroll
      for (int j = 0; j < 8; ++j) vT[c0 + j][srel] = (_Float16)src[j];
    }
    __syncthreads();
    if (wv < 4) {  // wave-uniform: EXEC all-ones inside
      v16h a = load_frag_a(&pH[l][kc], hi);
      v16h b = load_frag_b(&vT[wv * 16 + l][0], hi);
      acc = __builtin_amdgcn_wmma_f32_16x16x32_f16(false, a, false, b,
                                                   (short)0, acc, false, false);
    }
    __syncthreads();
  }
  if (wv < 4) {
#pragma unroll
    for (int r = 0; r < 8; ++r) {
      const int t = tBase + hi * 8 + r;
      attnH[t * kE + h * kD + wv * 16 + l] = (_Float16)acc[r];
    }
  }
}

// ---------------------------------------------------------------------------
extern "C" void kernel_launch(void* const* d_in, const int* in_sizes, int n_in,
                              void* d_out, int out_size, void* d_ws, size_t ws_size,
                              hipStream_t stream) {
  (void)in_sizes; (void)n_in; (void)out_size; (void)ws_size;
  const float* x   = (const float*)d_in[0];
  const float* Wq  = (const float*)d_in[1];
  const float* bq  = (const float*)d_in[2];
  const float* Wk  = (const float*)d_in[3];
  const float* bk  = (const float*)d_in[4];
  const float* Wv  = (const float*)d_in[5];
  const float* bv  = (const float*)d_in[6];
  const float* Wo  = (const float*)d_in[7];
  const float* bo  = (const float*)d_in[8];
  const float* wsv = (const float*)d_in[9];
  const float* bs  = (const float*)d_in[10];
  float* out = (float*)d_out;

  const int NX = kT * kE;   // 393216
  const int NW = kE * kE;   // 589824

  float* Q = (float*)d_ws;
  float* K = Q + NX;
  float* V = K + NX;
  _Float16* Xh    = (_Float16*)(V + NX);
  _Float16* Wqh   = Xh + NX;
  _Float16* Wkh   = Wqh + NW;
  _Float16* Wvh   = Wkh + NW;
  _Float16* Woh   = Wvh + NW;
  _Float16* attnH = Woh + NW;

  // pass 0: convert activations + weights to f16 (byte-exact async staging)
  cvt_f32_f16<<<(NX / 4 + 255) / 256, 256, 0, stream>>>(x, Xh, NX / 4);
  cvt_f32_f16<<<(NW / 4 + 255) / 256, 256, 0, stream>>>(Wq, Wqh, NW / 4);
  cvt_f32_f16<<<(NW / 4 + 255) / 256, 256, 0, stream>>>(Wk, Wkh, NW / 4);
  cvt_f32_f16<<<(NW / 4 + 255) / 256, 256, 0, stream>>>(Wv, Wvh, NW / 4);
  cvt_f32_f16<<<(NW / 4 + 255) / 256, 256, 0, stream>>>(Wo, Woh, NW / 4);

  dim3 gG(kT / 128, kE / 64);  // (4, 12)
  gemm_h_xwt_bias<<<gG, 256, 0, stream>>>(Xh, Wqh, bq, Q);
  gemm_h_xwt_bias<<<gG, 256, 0, stream>>>(Xh, Wkh, bk, K);
  gemm_h_xwt_bias<<<gG, 256, 0, stream>>>(Xh, Wvh, bv, V);
  attn_kernel<<<dim3(kT / 16, kH), 256, 0, stream>>>(Q, K, V, wsv, bs, attnH);
  gemm_h_xwt_bias<<<gG, 256, 0, stream>>>(attnH, Woh, bo, out);
}
